// GCN_11639361372218
// MI455X (gfx1250) — matrix operand order repaired and checked by reading, not compiled, not run on hardware
//
#include <hip/hip_runtime.h>
#include <math.h>
#include <stdint.h>

typedef __attribute__((ext_vector_type(16))) _Float16 v16h;
typedef __attribute__((ext_vector_type(8)))  _Float16 v8h;
typedef __attribute__((ext_vector_type(8)))  float    v8f;

// ---------------------------------------------------------------------------
// Helpers
// ---------------------------------------------------------------------------
__device__ __forceinline__ v16h load_frag(const _Float16* __restrict__ p0,
                                          const _Float16* __restrict__ p1) {
  v8h lo = *(const v8h*)p0;   // global_load_b128
  v8h hi = *(const v8h*)p1;   // global_load_b128
  v16h r;
#pragma unroll
  for (int i = 0; i < 8; ++i) { r[i] = lo[i]; r[8 + i] = hi[i]; }
  return r;
}

// ---------------------------------------------------------------------------
// Weight conversion: Wt[n][k] = (f16) W[k][n], rows n >= Norig zero-padded.
// ---------------------------------------------------------------------------
__global__ void cvt_w_kernel(const float* __restrict__ W, _Float16* __restrict__ Wt,
                             int K, int Norig, int Npad) {
  int i = blockIdx.x * blockDim.x + threadIdx.x;
  int total = Npad * K;
  if (i >= total) return;
  int n = i / K;
  int k = i - n * K;
  Wt[i] = (n < Norig) ? (_Float16)W[(size_t)k * Norig + n] : (_Float16)0.0f;
}

// x (f32) -> f16
__global__ void cvt_x_kernel(const float* __restrict__ x, _Float16* __restrict__ xh,
                             int total) {
  int i = blockIdx.x * blockDim.x + threadIdx.x;
  if (i < total) xh[i] = (_Float16)x[i];
}

// agg + bias, relu, convert to f16 for the next GEMM
__global__ void bias_relu_cvt_kernel(const float* __restrict__ agg,
                                     const float* __restrict__ bias,
                                     _Float16* __restrict__ outh,
                                     int total, int cmask) {
  int i = blockIdx.x * blockDim.x + threadIdx.x;
  if (i >= total) return;
  float v = agg[i] + bias[i & cmask];
  outh[i] = (_Float16)fmaxf(v, 0.0f);
}

// ---------------------------------------------------------------------------
// WMMA GEMM: C[M x Cstride] = A[M x K] @ Bt^T, Bt is [Npad x K] (row n = col n)
// 8 waves/block, each wave computes a 16 x (NT*16) slab.
// ---------------------------------------------------------------------------
template <int NT>
__global__ __launch_bounds__(256) void gemm_wmma_f16(
    const _Float16* __restrict__ A, const _Float16* __restrict__ Bt,
    float* __restrict__ C, int Mtiles, int K, int Cstride) {
  const int wave = threadIdx.x >> 5;
  const int lane = threadIdx.x & 31;
  const int mt = blockIdx.x * 8 + wave;
  if (mt >= Mtiles) return;                 // wave-uniform: EXEC stays all-1s
  const int lo16 = lane & 15;
  const int hi = lane >> 4;
  const int nb = blockIdx.y * (NT * 16);

  // A layout (16-bit A 16x32): lanes 0-15 hold K+0..7 / K+16..23,
  //                            lanes 16-31 hold K+8..15 / K+24..31 for row M=lane&15
  const _Float16* arow = A + (size_t)(mt * 16 + lo16) * K;
  const int aOff0 = hi ? 8 : 0;
  const int aOff1 = hi ? 24 : 16;
  // B layout: lane = column (mod 16); same K split as A (hi half-wave: K+8..15 / K+24..31)
  const int bOff = hi * 16;

  v8f zero = {};
  v8f acc[NT];
#pragma unroll
  for (int t = 0; t < NT; ++t) acc[t] = zero;

  for (int k = 0; k < K; k += 32) {
    v16h afrag = load_frag(arow + k + aOff0, arow + k + aOff1);
#pragma unroll
    for (int t = 0; t < NT; ++t) {
      const _Float16* brow = Bt + (size_t)(nb + t * 16 + lo16) * K + k + bOff;
      v16h bfrag = load_frag(brow - bOff + aOff0, brow - bOff + aOff1);
      acc[t] = __builtin_amdgcn_wmma_f32_16x16x32_f16(
          false, afrag, false, bfrag, (short)0, acc[t], false, false);
    }
  }

  // C/D layout: lanes 0-15 -> M = r, lanes 16-31 -> M = 8 + r; N = lane & 15
  const int mrow = mt * 16 + hi * 8;
#pragma unroll
  for (int t = 0; t < NT; ++t) {
    float* cp = C + (size_t)mrow * Cstride + nb + t * 16 + lo16;
#pragma unroll
    for (int r = 0; r < 8; ++r) cp[(size_t)r * Cstride] = acc[t][r];
  }
}

// ---------------------------------------------------------------------------
// Edge aggregation, 256 channels: one wave per edge, 2 x float4 per lane.
// agg[dst] += ew * h[src]   (hardware fp32 atomics, no return)
// ---------------------------------------------------------------------------
__global__ __launch_bounds__(256) void aggregate256_kernel(
    const float* __restrict__ h, const int* __restrict__ src,
    const int* __restrict__ dst, const float* __restrict__ ew,
    float* __restrict__ agg, int E) {
  const int wid = (blockIdx.x * blockDim.x + threadIdx.x) >> 5;
  const int lane = threadIdx.x & 31;
  if (wid >= E) return;
  const int s = src[wid];
  const int d = dst[wid];
  const float w = ew[wid];
  const float4* hr = (const float4*)(h + (size_t)s * 256);
  float* ar = agg + (size_t)d * 256;
#pragma unroll
  for (int i = 0; i < 2; ++i) {
    const int q = lane + i * 32;
    float4 v = hr[q];
    unsafeAtomicAdd(ar + q * 4 + 0, v.x * w);
    unsafeAtomicAdd(ar + q * 4 + 1, v.y * w);
    unsafeAtomicAdd(ar + q * 4 + 2, v.z * w);
    unsafeAtomicAdd(ar + q * 4 + 3, v.w * w);
  }
}

// Edge aggregation, 40 channels (h stored with stride 48)
__global__ __launch_bounds__(256) void aggregate40_kernel(
    const float* __restrict__ h, const int* __restrict__ src,
    const int* __restrict__ dst, const float* __restrict__ ew,
    float* __restrict__ agg, int E) {
  const int wid = (blockIdx.x * blockDim.x + threadIdx.x) >> 5;
  const int lane = threadIdx.x & 31;
  if (wid >= E) return;
  const int s = src[wid];
  const int d = dst[wid];
  const float w = ew[wid];
  const float* hr = h + (size_t)s * 48;
  float* ar = agg + (size_t)d * 40;
  unsafeAtomicAdd(ar + lane, hr[lane] * w);
  if (lane < 8) unsafeAtomicAdd(ar + 32 + lane, hr[32 + lane] * w);
}

// ---------------------------------------------------------------------------
// Final: out = log_softmax(agg + b3) over 40 channels, one wave per node.
// ---------------------------------------------------------------------------
__global__ __launch_bounds__(256) void logsoftmax40_kernel(
    const float* __restrict__ agg, const float* __restrict__ b,
    float* __restrict__ out, int Nn) {
  const int node = (blockIdx.x * blockDim.x + threadIdx.x) >> 5;
  const int lane = threadIdx.x & 31;
  if (node >= Nn) return;
  const float* a = agg + (size_t)node * 40;
  float v0 = a[lane] + b[lane];
  float v1 = (lane < 8) ? (a[32 + lane] + b[32 + lane]) : -INFINITY;
  float m = fmaxf(v0, v1);
#pragma unroll
  for (int o = 16; o >= 1; o >>= 1) m = fmaxf(m, __shfl_xor(m, o, 32));
  float s = __expf(v0 - m) + ((lane < 8) ? __expf(v1 - m) : 0.0f);
#pragma unroll
  for (int o = 16; o >= 1; o >>= 1) s += __shfl_xor(s, o, 32);
  float lse = m + __logf(s);
  float* op = out + (size_t)node * 40;
  op[lane] = v0 - lse;
  if (lane < 8) op[32 + lane] = v1 - lse;
}

// ---------------------------------------------------------------------------
// Launch
// ---------------------------------------------------------------------------
extern "C" void kernel_launch(void* const* d_in, const int* in_sizes, int n_in,
                              void* d_out, int out_size, void* d_ws, size_t ws_size,
                              hipStream_t stream) {
  const float* x  = (const float*)d_in[0];
  const int*   ei = (const int*)d_in[1];
  const float* ew = (const float*)d_in[2];
  const float* W1 = (const float*)d_in[3];
  const float* b1 = (const float*)d_in[4];
  const float* W2 = (const float*)d_in[5];
  const float* b2 = (const float*)d_in[6];
  const float* W3 = (const float*)d_in[7];
  const float* b3 = (const float*)d_in[8];
  float* out = (float*)d_out;

  const int INC = 128, HID = 256, OUTP = 48, OUTC = 40;
  const int Nn = in_sizes[0] / INC;     // 50000
  const int E  = in_sizes[2];           // 800000
  const int* src = ei;
  const int* dst = ei + E;

  uint8_t* ws = (uint8_t*)d_ws;
  size_t off = 0;
  _Float16* ah  = (_Float16*)(ws + off); off += (size_t)Nn * HID * sizeof(_Float16);
  float*    hbuf = (float*)(ws + off);   off += (size_t)Nn * HID * sizeof(float);
  float*    agg  = (float*)(ws + off);   off += (size_t)Nn * HID * sizeof(float);
  _Float16* w1t = (_Float16*)(ws + off); off += (size_t)HID * INC * sizeof(_Float16);
  _Float16* w2t = (_Float16*)(ws + off); off += (size_t)HID * HID * sizeof(_Float16);
  _Float16* w3t = (_Float16*)(ws + off); off += (size_t)OUTP * HID * sizeof(_Float16);

  const int TB = 256;
  const int Mtiles = Nn / 16;                        // 3125 (N is a multiple of 16)
  const dim3 gemm_blk((Mtiles + 7) / 8, HID / 128);  // 16x128 slab per wave (NT=8)
  const int edge_blocks = (E + 7) / 8;               // one wave per edge

  // --- prep: weights -> transposed f16, x -> f16 ---
  cvt_w_kernel<<<(HID * INC + TB - 1) / TB, TB, 0, stream>>>(W1, w1t, INC, HID, HID);
  cvt_w_kernel<<<(HID * HID + TB - 1) / TB, TB, 0, stream>>>(W2, w2t, HID, HID, HID);
  cvt_w_kernel<<<(OUTP * HID + TB - 1) / TB, TB, 0, stream>>>(W3, w3t, HID, OUTC, OUTP);
  cvt_x_kernel<<<((Nn * INC) + TB - 1) / TB, TB, 0, stream>>>(x, ah, Nn * INC);

  // --- layer 1 ---
  gemm_wmma_f16<8><<<gemm_blk, TB, 0, stream>>>(ah, w1t, hbuf, Mtiles, INC, HID);
  hipMemsetAsync(agg, 0, (size_t)Nn * HID * sizeof(float), stream);
  aggregate256_kernel<<<edge_blocks, TB, 0, stream>>>(hbuf, src, dst, ew, agg, E);
  bias_relu_cvt_kernel<<<((Nn * HID) + TB - 1) / TB, TB, 0, stream>>>(
      agg, b1, ah, Nn * HID, HID - 1);

  // --- layer 2 ---
  gemm_wmma_f16<8><<<gemm_blk, TB, 0, stream>>>(ah, w2t, hbuf, Mtiles, HID, HID);
  hipMemsetAsync(agg, 0, (size_t)Nn * HID * sizeof(float), stream);
  aggregate256_kernel<<<edge_blocks, TB, 0, stream>>>(hbuf, src, dst, ew, agg, E);
  bias_relu_cvt_kernel<<<((Nn * HID) + TB - 1) / TB, TB, 0, stream>>>(
      agg, b2, ah, Nn * HID, HID - 1);

  // --- layer 3 (N padded 40 -> 48, zero weight rows) ---
  gemm_wmma_f16<3><<<dim3((Mtiles + 7) / 8, 1), TB, 0, stream>>>(
      ah, w3t, hbuf, Mtiles, HID, OUTP);
  hipMemsetAsync(agg, 0, (size_t)Nn * OUTC * sizeof(float), stream);
  aggregate40_kernel<<<edge_blocks, TB, 0, stream>>>(hbuf, src, dst, ew, agg, E);
  logsoftmax40_kernel<<<(Nn + 7) / 8, TB, 0, stream>>>(agg, b3, out, Nn);
}
